// Net3_16587163698029
// MI455X (gfx1250) — matrix operand order repaired and checked by reading, not compiled, not run on hardware
//
#include <hip/hip_runtime.h>
#include <hip/hip_bf16.h>

#define NGRAPH 1000
#define NNODES 100
#define EPG    1000
#define FIN    14
#define H      128
#define KP1    80
#define KP2    64
#define MPAD   112            // 7 M-tiles of 16
#define NEDGES_TOTAL (NGRAPH * EPG)

typedef __attribute__((ext_vector_type(16))) _Float16 v16h;
typedef __attribute__((ext_vector_type(8)))  _Float16 v8h;
typedef __attribute__((ext_vector_type(8)))  float    v8f;

static __device__ __forceinline__ v16h cat8(v8h lo, v8h hi) {
    return __builtin_shufflevector(lo, hi, 0, 1, 2, 3, 4, 5, 6, 7,
                                           8, 9, 10, 11, 12, 13, 14, 15);
}

__global__ __launch_bounds__(256) void gnn_graph_kernel(
    const float* __restrict__ x,      const int* __restrict__ edge_index,
    const float* __restrict__ Wrel1,  const float* __restrict__ Wroot1,
    const float* __restrict__ b1,     const float* __restrict__ p1,
    const float* __restrict__ Wrel2,  const float* __restrict__ Wroot2,
    const float* __restrict__ b2,     const float* __restrict__ p2,
    const float* __restrict__ l1w,    const float* __restrict__ l1b,
    const float* __restrict__ l2w,    const float* __restrict__ l2b,
    const float* __restrict__ l3w,    const float* __restrict__ l3b,
    float* __restrict__ out)
{
    __shared__ float    s_agg[MPAD * H];                 // 57344 B  f32 scatter target (ds_add_f32)
    __shared__ alignas(16) _Float16 s_agg16[MPAD * H];   // 28672 B  f16 A-stage (agg half); also A1 stage
    __shared__ alignas(16) _Float16 s_h16[MPAD * H];     // 28672 B  f16 node features (A root half)
    __shared__ alignas(16) _Float16 s_B1t[H * 32];       //  8192 B  col-major [Wrel1;Wroot1;0]
    __shared__ alignas(16) _Float16 s_B2t[H * 256];      // 65536 B  col-major [Wrel2;Wroot2]
    __shared__ float    s_x[MPAD * 16];                  //  7168 B  input feats, stride 16, padded
    __shared__ unsigned short s_src[EPG];
    __shared__ unsigned short s_dst[EPG];
    __shared__ float    s_score[MPAD];
    __shared__ float    s_mask1[MPAD];
    __shared__ float    s_mask2[MPAD];
    __shared__ float    s_vec[4 * H];                    // b1 | b2 | p1 | p2
    __shared__ float    s_norm[2];
    __shared__ float    s_z[2 * H];
    __shared__ float    s_m1[H];
    __shared__ float    s_m2[32];

    const int g    = blockIdx.x;
    const int tid  = threadIdx.x;
    const int lane = tid & 31;
    const int wv   = tid >> 5;
    const int nlo  = lane & 15;      // M (A) / N (B,C,D) index within tile
    const int lhi  = lane >> 4;      // lane half selector
    const int nodeBase = g * NNODES;
    const int edgeBase = g * EPG;

    // ---------------- Phase 0: stage graph data + weights ----------------
    for (int i = tid; i < MPAD * 16; i += 256) {
        int r = i >> 4, c = i & 15;
        float v = 0.f;
        if (r < NNODES && c < FIN) v = x[(nodeBase + r) * FIN + c];
        s_x[i] = v;
    }
    for (int e = tid; e < EPG; e += 256) {
        s_src[e] = (unsigned short)(edge_index[edgeBase + e] - nodeBase);
        s_dst[e] = (unsigned short)(edge_index[NEDGES_TOTAL + edgeBase + e] - nodeBase);
    }
    for (int i = tid; i < 32 * H; i += 256) {        // B1t[col][k] = [Wrel1(14);Wroot1(14);0(4)]
        int k = i >> 7, col = i & 127;
        float v = 0.f;
        if (k < FIN)          v = Wrel1[k * H + col];
        else if (k < 2 * FIN) v = Wroot1[(k - FIN) * H + col];
        s_B1t[col * 32 + k] = (_Float16)v;
    }
    for (int i = tid; i < 256 * H; i += 256) {       // B2t[col][k] = [Wrel2(128);Wroot2(128)]
        int k = i >> 7, col = i & 127;
        float v = (k < H) ? Wrel2[k * H + col] : Wroot2[(k - H) * H + col];
        s_B2t[col * 256 + k] = (_Float16)v;
    }
    for (int i = tid; i < 4 * H; i += 256) {
        int which = i >> 7, j = i & 127;
        const float* src = (which == 0) ? b1 : (which == 1) ? b2 : (which == 2) ? p1 : p2;
        s_vec[i] = src[j];
    }
    for (int i = tid; i < MPAD * 16; i += 256) s_agg[i] = 0.f;   // conv1 agg region
    for (int i = tid; i < 2 * H; i += 256) s_z[i] = 0.f;
    if (tid < 2) {
        const float* p = (tid == 0) ? p1 : p2;
        float s = 0.f;
        for (int j = 0; j < H; ++j) s += p[j] * p[j];
        s_norm[tid] = sqrtf(s);
    }
    __syncthreads();

    // ---------------- Phase 1: conv1 edge scatter (ds_add_f32) ----------------
    for (int e = tid; e < EPG; e += 256) {
        int s = s_src[e], d = s_dst[e];
        #pragma unroll
        for (int f = 0; f < FIN; ++f)
            atomicAdd(&s_agg[d * 16 + f], s_x[s * 16 + f]);
    }
    __syncthreads();

    // ---------------- Phase 1b: build f16 A1 = [agg1 | x | 0]  (112 x 32, in s_agg16) ------
    for (int i = tid; i < MPAD * 32; i += 256) {
        int r = i >> 5, c = i & 31;
        float v;
        if (c < FIN)          v = s_agg[r * 16 + c];
        else if (c < 2 * FIN) v = s_x[r * 16 + (c - FIN)];
        else                  v = 0.f;
        s_agg16[i] = (_Float16)v;
    }
    __syncthreads();

    // ---------------- Phase 2: conv1 WMMA: h = relu(A1 @ B1 + b1) ----------------
    {
        v8f acc[7];
        #pragma unroll
        for (int i = 0; i < 7; ++i) {
            int t = wv * 7 + i, mt = t >> 3, nt = t & 7;
            int r = mt * 16 + nlo, col = nt * 16 + nlo;
            v16h a = cat8(*(const v8h*)&s_agg16[r * 32 + lhi * 8],
                          *(const v8h*)&s_agg16[r * 32 + 16 + lhi * 8]);
            v16h b = cat8(*(const v8h*)&s_B1t[col * 32 + lhi * 8],
                          *(const v8h*)&s_B1t[col * 32 + 16 + lhi * 8]);
            v8f c = {};
            acc[i] = __builtin_amdgcn_wmma_f32_16x16x32_f16(
                false, a, false, b, (short)0, c, false, false);
        }
        #pragma unroll
        for (int i = 0; i < 7; ++i) {
            int t = wv * 7 + i, mt = t >> 3, nt = t & 7;
            #pragma unroll
            for (int v = 0; v < 8; ++v) {
                int r = mt * 16 + lhi * 8 + v;
                int c = nt * 16 + nlo;
                float val = acc[i][v] + s_vec[c];               // + b1
                s_h16[r * H + c] = (_Float16)((r < NNODES) ? fmaxf(val, 0.f) : 0.f);
            }
        }
    }
    __syncthreads();

    // ---------------- Phase 3: TopK pool 1 (all nodes alive) ----------------
    if (tid < MPAD) {
        float sc = 0.f;
        if (tid < NNODES) {
            float d = 0.f;
            for (int j = 0; j < H; ++j) d += (float)s_h16[tid * H + j] * s_vec[2 * H + j];
            sc = tanhf(d / s_norm[0]);
        }
        s_score[tid] = sc;
    }
    __syncthreads();
    if (tid < MPAD) {
        float mk = 0.f;
        if (tid < NNODES) {
            float si = s_score[tid];
            int rank = 0;
            for (int j = 0; j < NNODES; ++j) {
                float sj = s_score[j];
                if (sj > si || (sj == si && j < tid)) rank++;
            }
            mk = (rank < KP1) ? 1.f : 0.f;
        }
        s_mask1[tid] = mk;
    }
    __syncthreads();
    for (int i = tid; i < MPAD * H; i += 256) {      // h *= s*mask ; zero agg for conv2
        int r = i >> 7;
        s_h16[i] = (_Float16)((float)s_h16[i] * (s_score[r] * s_mask1[r]));
        s_agg[i] = 0.f;
    }
    __syncthreads();

    // ---------------- Phase 4: readout 1 + conv2 edge scatter ----------------
    if (tid < H) {
        float sum = 0.f, mx = -__builtin_inff();
        for (int i = 0; i < NNODES; ++i) {
            float v = (float)s_h16[i * H + tid];
            sum += v;
            if (s_mask1[i] > 0.f) mx = fmaxf(mx, v);
        }
        s_z[tid]     += sum / (float)KP1;
        s_z[H + tid] += mx;
    }
    // msg = h[src]*m1[src]*m1[dst]; h already zeroed at dead src, so gate on dst only
    for (int idx = tid; idx < EPG * 8; idx += 256) {
        int e  = idx >> 3;
        int fc = (idx & 7) << 4;
        int d  = s_dst[e];
        if (s_mask1[d] > 0.f) {
            int s = s_src[e];
            #pragma unroll
            for (int f = 0; f < 16; ++f)
                atomicAdd(&s_agg[d * H + fc + f], (float)s_h16[s * H + fc + f]);
        }
    }
    __syncthreads();

    // ---------------- Phase 4b: f16 stage of agg2 ----------------
    for (int i = tid; i < MPAD * H; i += 256)
        s_agg16[i] = (_Float16)s_agg[i];
    __syncthreads();

    // ---------------- Phase 5: conv2 WMMA: h2 = relu([agg2|h] @ [Wrel2;Wroot2] + b2) -------
    {
        v8f acc[7];
        #pragma unroll
        for (int i = 0; i < 7; ++i) { v8f c = {}; acc[i] = c; }
        for (int kb = 0; kb < 2 * H; kb += 32) {
            const _Float16* Ab = (kb < H) ? s_agg16 : s_h16;   // uniform per K-step
            int aoff = (kb < H) ? kb : (kb - H);
            #pragma unroll
            for (int i = 0; i < 7; ++i) {
                int t = wv * 7 + i, mt = t >> 3, nt = t & 7;
                int r = mt * 16 + nlo, col = nt * 16 + nlo;
                v16h a = cat8(*(const v8h*)&Ab[r * H + aoff + lhi * 8],
                              *(const v8h*)&Ab[r * H + aoff + 16 + lhi * 8]);
                v16h b = cat8(*(const v8h*)&s_B2t[col * 256 + kb + lhi * 8],
                              *(const v8h*)&s_B2t[col * 256 + kb + 16 + lhi * 8]);
                acc[i] = __builtin_amdgcn_wmma_f32_16x16x32_f16(
                    false, a, false, b, (short)0, acc[i], false, false);
            }
        }
        __syncthreads();   // all A reads complete before overwriting s_h16
        #pragma unroll
        for (int i = 0; i < 7; ++i) {
            int t = wv * 7 + i, mt = t >> 3, nt = t & 7;
            #pragma unroll
            for (int v = 0; v < 8; ++v) {
                int r = mt * 16 + lhi * 8 + v;
                int c = nt * 16 + nlo;
                float val = acc[i][v] + s_vec[H + c];           // + b2
                s_h16[r * H + c] = (_Float16)((r < NNODES) ? fmaxf(val, 0.f) : 0.f);
            }
        }
    }
    __syncthreads();

    // ---------------- Phase 6: TopK pool 2 (alive = m1) ----------------
    if (tid < MPAD) {
        float sc = 0.f;
        if (tid < NNODES) {
            float d = 0.f;
            for (int j = 0; j < H; ++j) d += (float)s_h16[tid * H + j] * s_vec[3 * H + j];
            sc = tanhf(d / s_norm[1]);
        }
        s_score[tid] = sc;
    }
    __syncthreads();
    if (tid < MPAD) {
        float mk = 0.f;
        if (tid < NNODES && s_mask1[tid] > 0.f) {
            float si = s_score[tid];
            int rank = 0;
            for (int j = 0; j < NNODES; ++j) {
                if (s_mask1[j] > 0.f) {
                    float sj = s_score[j];
                    if (sj > si || (sj == si && j < tid)) rank++;
                }
            }
            mk = (rank < KP2) ? 1.f : 0.f;
        }
        s_mask2[tid] = mk;
    }
    __syncthreads();
    for (int i = tid; i < MPAD * H; i += 256) {
        int r = i >> 7;
        s_h16[i] = (_Float16)((float)s_h16[i] * (s_score[r] * s_mask2[r]));
    }
    __syncthreads();

    // ---------------- Phase 7: readout 2 ----------------
    if (tid < H) {
        float sum = 0.f, mx = -__builtin_inff();
        for (int i = 0; i < NNODES; ++i) {
            float v = (float)s_h16[i * H + tid];
            sum += v;
            if (s_mask2[i] > 0.f) mx = fmaxf(mx, v);
        }
        s_z[tid]     += sum / (float)KP2;
        s_z[H + tid] += mx;
    }
    __syncthreads();

    // ---------------- Phase 8: MLP head + log_softmax ----------------
    if (tid < H) {
        float a = l1b[tid];
        for (int j = 0; j < 2 * H; ++j) a += s_z[j] * l1w[j * H + tid];
        s_m1[tid] = fmaxf(a, 0.f);
    }
    __syncthreads();
    if (tid < 32) {
        float a = l2b[tid];
        for (int j = 0; j < H; ++j) a += s_m1[j] * l2w[j * 32 + tid];
        s_m2[tid] = fmaxf(a, 0.f);
    }
    __syncthreads();
    if (tid == 0) {
        float z0 = l3b[0], z1 = l3b[1];
        for (int j = 0; j < 32; ++j) {
            z0 += s_m2[j] * l3w[j * 2 + 0];
            z1 += s_m2[j] * l3w[j * 2 + 1];
        }
        float m   = fmaxf(z0, z1);
        float lse = m + logf(expf(z0 - m) + expf(z1 - m));
        out[g * 2 + 0] = z0 - lse;
        out[g * 2 + 1] = z1 - lse;
    }
}

extern "C" void kernel_launch(void* const* d_in, const int* in_sizes, int n_in,
                              void* d_out, int out_size, void* d_ws, size_t ws_size,
                              hipStream_t stream) {
    (void)in_sizes; (void)n_in; (void)d_ws; (void)ws_size; (void)out_size;
    gnn_graph_kernel<<<NGRAPH, 256, 0, stream>>>(
        (const float*)d_in[0],  (const int*)d_in[1],
        (const float*)d_in[2],  (const float*)d_in[3],  (const float*)d_in[4],
        (const float*)d_in[5],
        (const float*)d_in[6],  (const float*)d_in[7],  (const float*)d_in[8],
        (const float*)d_in[9],
        (const float*)d_in[10], (const float*)d_in[11],
        (const float*)d_in[12], (const float*)d_in[13],
        (const float*)d_in[14], (const float*)d_in[15],
        (float*)d_out);
}
